// LinearAttentionSVAR_61040075210835
// MI455X (gfx1250) — matrix operand order repaired and checked by reading, not compiled, not run on hardware
//
#include <hip/hip_runtime.h>
#include <hip/hip_bf16.h>

// Problem constants (from reference): B=2, L=2048, D=1024, H=16, d=64
#define BSZ 2
#define LSEQ 2048
#define DMODEL 1024
#define NHEAD 16
#define DHEAD 64
#define MTOT (BSZ * LSEQ)       // 4096
#define NCHUNK (LSEQ / 64)      // 32 chunks per sequence
#define NBH (BSZ * NHEAD)       // 32

typedef __bf16 bf16_t;
typedef __attribute__((ext_vector_type(16))) __bf16 v16bf;
typedef __attribute__((ext_vector_type(8)))  float  v8f;
typedef unsigned int u32x4 __attribute__((ext_vector_type(4)));
typedef unsigned int u32x8 __attribute__((ext_vector_type(8)));

// ---------------------------------------------------------------------------
// WMMA operand packing helpers (CDNA5 wave32 layouts, CDNA5 ISA)
// A-matrix 16x32 bf16: lane (l&15)=M; lanes<16 hold K={0..7,16..23},
//                      lanes>=16 hold K={8..15,24..31}.
// B-matrix 32x16 bf16: lane (l&15)=N; lanes<16 hold K={0..7,16..23},
//                      lanes>=16 hold K={8..15,24..31}.
// C/D 16x16 f32: element r -> M = r + 8*(l>=16), N = l&15.
// ---------------------------------------------------------------------------
__device__ __forceinline__ v16bf load_A_rm(const bf16_t* __restrict__ base,
                                           int stride, int lane) {
  const int m = lane & 15;
  const int h8 = (lane >> 4) ? 8 : 0;
  const bf16_t* row = base + (size_t)m * stride + h8;
  v16bf a;
#pragma unroll
  for (int j = 0; j < 8; ++j) a[j] = row[j];
#pragma unroll
  for (int j = 0; j < 8; ++j) a[8 + j] = row[16 + j];
  return a;
}

__device__ __forceinline__ v16bf load_B_t(const bf16_t* __restrict__ base,
                                          int stride, int lane) {
  // B[k][n] = base[n*stride + k]
  const int n = lane & 15;
  const int h8 = (lane >> 4) ? 8 : 0;
  const bf16_t* row = base + (size_t)n * stride + h8;
  v16bf b;
#pragma unroll
  for (int j = 0; j < 8; ++j) b[j] = row[j];
#pragma unroll
  for (int j = 0; j < 8; ++j) b[8 + j] = row[16 + j];
  return b;
}

__device__ __forceinline__ v8f wmma_bf16(v16bf a, v16bf b, v8f c) {
  return __builtin_amdgcn_wmma_f32_16x16x32_bf16(
      /*neg_a=*/false, a, /*neg_b=*/false, b,
      /*c_mod=*/(short)0, c, /*reuse_a=*/false, /*reuse_b=*/false);
}

// Async global->LDS copy of 16 bytes (GLOBAL_LOAD_ASYNC_TO_LDS_B128, ASYNCcnt).
__device__ __forceinline__ void async_copy_b128(const void* gsrc, void* ldst) {
  unsigned lds_off = (unsigned)(size_t)ldst;            // low 32 bits = LDS offset
  unsigned long long ga = (unsigned long long)(size_t)gsrc;
  asm volatile("global_load_async_to_lds_b128 %0, %1, off"
               :: "v"(lds_off), "v"(ga) : "memory");
}

__device__ __forceinline__ void wait_async0() {
  asm volatile("s_wait_asynccnt 0x0" ::: "memory");
}

// ---------------------------------------------------------------------------
// Tensor Data Mover: DMA a 2D tile (tile_w bf16 x tile_h rows, row stride
// row_stride elements) from global memory into LDS (row-major, packed).
// D# descriptor per CDNA5 ISA; groups 2/3 zero (2D tile,
// tile_dim2 = 0). Issue from ONE wave only; completion via TENSORcnt.
// ---------------------------------------------------------------------------
__device__ __forceinline__ void tdm_load_tile_2d(const void* gsrc, void* ldst,
                                                 unsigned tile_w, unsigned tile_h,
                                                 unsigned row_stride) {
  unsigned long long ga = (unsigned long long)(size_t)gsrc;
  u32x4 g0;
  g0[0] = 1u;                                   // count=1 (valid user descriptor)
  g0[1] = (unsigned)(size_t)ldst;               // lds_addr (bytes)
  g0[2] = (unsigned)ga;                         // global_addr[31:0]
  g0[3] = (unsigned)(ga >> 32) | (2u << 30);    // global_addr[56:32] | type=2
  u32x8 g1;
  g1[0] = 0x00010000u;                          // wg_mask=0, data_size=1 (2 bytes)
  g1[1] = (row_stride & 0xFFFFu) << 16;         // tensor_dim0[15:0] @ bits 63:48
  g1[2] = ((row_stride >> 16) & 0xFFFFu)        // tensor_dim0[31:16]
        | ((1024u & 0xFFFFu) << 16);            // tensor_dim1[15:0]
  g1[3] = (1024u >> 16)                         // tensor_dim1[31:16]
        | (tile_w << 16);                       // tile_dim0
  g1[4] = tile_h;                               // tile_dim1 (tile_dim2 = 0)
  g1[5] = row_stride;                           // tensor_dim0_stride[31:0]
  g1[6] = (unsigned)0;                          // stride hi / dim1_stride lo
  g1[7] = 0u;
  u32x4 gz = {0u, 0u, 0u, 0u};                  // groups 2/3: unused dims
  asm volatile("tensor_load_to_lds %0, %1, %2, %3"
               :: "s"(g0), "s"(g1), "s"(gz), "s"(gz) : "memory");
}

__device__ __forceinline__ void wait_tensor0() {
  asm volatile("s_wait_tensorcnt 0x0" ::: "memory");
}

// ---------------------------------------------------------------------------
// Kernel 1: f32 -> bf16 cast
// ---------------------------------------------------------------------------
__global__ void cast_f32_bf16(const float* __restrict__ in,
                              bf16_t* __restrict__ out, int n) {
  int i = blockIdx.x * blockDim.x + threadIdx.x;
  if (i < n) out[i] = (bf16_t)in[i];
}

// ---------------------------------------------------------------------------
// Kernel 2: Bm[h] = tril(Ltri[h]) @ tril(Ltri[h])^T   (16 x 64x64, f32)
// ---------------------------------------------------------------------------
__global__ void build_bm(const float* __restrict__ Ltri, float* __restrict__ Bm) {
  const int h = blockIdx.x;
  const float* Lh = Ltri + (size_t)h * DHEAD * DHEAD;
  float* Bh = Bm + (size_t)h * DHEAD * DHEAD;
  for (int i = threadIdx.x; i < DHEAD * DHEAD; i += blockDim.x) {
    const int e = i >> 6, dd = i & 63;
    const int kmax = e < dd ? e : dd;
    float s = 0.f;
    for (int k = 0; k <= kmax; ++k) s += Lh[e * DHEAD + k] * Lh[dd * DHEAD + k];
    Bh[i] = s;
  }
}

// ---------------------------------------------------------------------------
// Kernel 3: fold head-mixing into output projection:
// Wc2[n][h*64+dd] = sum_e Wc[n][h*64+e] * Bm[h][e][dd]   (bf16 out)
// ---------------------------------------------------------------------------
__global__ void fold_wc(const float* __restrict__ Wc, const float* __restrict__ Bm,
                        bf16_t* __restrict__ Wc2) {
  int idx = blockIdx.x * blockDim.x + threadIdx.x;
  if (idx >= DMODEL * DMODEL) return;
  const int n = idx >> 10;
  const int col = idx & (DMODEL - 1);
  const int h = col >> 6;
  const int dd = col & 63;
  const float* wrow = Wc + (size_t)n * DMODEL + h * DHEAD;
  const float* bh = Bm + (size_t)h * DHEAD * DHEAD;
  float s = 0.f;
#pragma unroll 8
  for (int e = 0; e < DHEAD; ++e) s += wrow[e] * bh[e * DHEAD + dd];
  Wc2[idx] = (bf16_t)s;
}

// ---------------------------------------------------------------------------
// Kernel 4: pipelined WMMA GEMM  C = A @ W^T
//   Block: 256 threads (8 waves) -> 128(M) x 64(N) output tile.
//   Per k-step (32): shared 64x32 weight slice double-buffered in LDS via the
//   TENSOR DATA MOVER (one tensor_load_to_lds per step, issued by wave 0,
//   TENSORcnt-tracked), reused by all 8 waves; each wave register-blocks
//   1x4 16x16 tiles (A operand reused 4x).
// ---------------------------------------------------------------------------
template <bool PROJ>
__global__ __launch_bounds__(256) void gemm_wmma(const bf16_t* __restrict__ Ain,
                                                 const bf16_t* __restrict__ W,
                                                 void* __restrict__ OutP) {
  __shared__ bf16_t Bt[2][64 * 32];       // double-buffered weight slice (8 KB)

  const int tid = threadIdx.x;
  const int lane = tid & 31;
  const int wave = tid >> 5;
  const int bm = blockIdx.x >> 4;         // 32 M-blocks of 128 rows
  const int bn = blockIdx.x & 15;         // 16 N-blocks of 64 cols
  const int mbase = bm * 128 + wave * 16;
  const int nblock = bn * 64;

  const bf16_t* gW = W + (size_t)nblock * DMODEL;   // 64 weight rows, stride D

  // Prologue: TDM-stage k-slice 0 into buffer 0.
  if (wave == 0) {
    tdm_load_tile_2d(gW, &Bt[0][0], 32, 64, DMODEL);
    wait_tensor0();
  }
  __syncthreads();

  v8f acc[4] = {{}, {}, {}, {}};
  const int NK = DMODEL / 32;             // 32 k-steps
  for (int i = 0; i < NK; ++i) {
    if (i + 1 < NK && wave == 0) {        // TDM-prefetch next slice
      tdm_load_tile_2d(gW + (i + 1) * 32, &Bt[(i + 1) & 1][0], 32, 64, DMODEL);
    }
    const bf16_t* bb = &Bt[i & 1][0];
    v16bf a = load_A_rm(Ain + (size_t)mbase * DMODEL + i * 32, DMODEL, lane);
#pragma unroll
    for (int tn = 0; tn < 4; ++tn) {
      v16bf b = load_B_t(bb + (tn * 16) * 32, 32, lane);
      acc[tn] = wmma_bf16(a, b, acc[tn]);
    }
    if (wave == 0) {
      wait_tensor0();                     // next buffer landed in LDS
    }
    __syncthreads();                      // all waves done reading current buffer
  }

  const int nn = lane & 15;
  const int mofs = (lane >> 4) ? 8 : 0;
#pragma unroll
  for (int tn = 0; tn < 4; ++tn) {
    const int n = nblock + tn * 16 + nn;
    if (PROJ) {
      bf16_t* Out = (bf16_t*)OutP;
      const int h = n >> 6;
      const int dd = n & 63;
#pragma unroll
      for (int r = 0; r < 8; ++r) {
        const int m = mbase + r + mofs;
        const int bb2 = m >> 11;
        const int l = m & (LSEQ - 1);
        Out[(((size_t)bb2 * NHEAD + h) * LSEQ + l) * DHEAD + dd] = (bf16_t)acc[tn][r];
      }
    } else {
      float* Out = (float*)OutP;
#pragma unroll
      for (int r = 0; r < 8; ++r) {
        const int m = mbase + r + mofs;
        Out[(size_t)m * DMODEL + n] = acc[tn][r];
      }
    }
  }
}

// ---------------------------------------------------------------------------
// Kernel 5a: per-chunk K^T @ V (fully parallel over B*H*NCHUNK blocks).
// KVc[bhc][k][n] = sum_{j<64} K[l0+j][k] * V[l0+j][n]   (f32)
// ---------------------------------------------------------------------------
__global__ __launch_bounds__(256) void chunk_kv(const bf16_t* __restrict__ Kg,
                                                const bf16_t* __restrict__ Vg,
                                                float* __restrict__ KVc) {
  __shared__ bf16_t Kt[DHEAD * DHEAD];      // Kt[k][j] = K[l0+j][k]
  __shared__ bf16_t Vt[DHEAD * DHEAD];      // Vt[n][j] = V[l0+j][n]

  const int bhc = blockIdx.x;               // bh*NCHUNK + c
  const int bh = bhc / NCHUNK;
  const int c = bhc % NCHUNK;
  const int l0 = c * 64;
  const size_t base = (size_t)bh * LSEQ * DHEAD;
  const bf16_t* Km = Kg + base;
  const bf16_t* V = Vg + base;

  const int tid = threadIdx.x;
  const int lane = tid & 31;
  const int wave = tid >> 5;

  for (int i = tid; i < DHEAD * DHEAD; i += 256) {
    const int row = i >> 6, col = i & 63;
    Kt[i] = Km[(size_t)(l0 + col) * DHEAD + row];
    Vt[i] = V[(size_t)(l0 + col) * DHEAD + row];
  }
  __syncthreads();

  float* outc = KVc + (size_t)bhc * DHEAD * DHEAD;
  for (int t = wave; t < 16; t += 8) {
    const int ti = t >> 2, tj = t & 3;      // ti: k-block, tj: n-block
    v8f su = {};
    for (int k = 0; k < DHEAD; k += 32) {   // contraction over chunk rows j
      v16bf a = load_A_rm(Kt + (size_t)(ti * 16) * DHEAD + k, DHEAD, lane);
      v16bf b = load_B_t(Vt + (size_t)(tj * 16) * DHEAD + k, DHEAD, lane);
      su = wmma_bf16(a, b, su);
    }
    const int n = lane & 15;
    const int mofs = (lane >> 4) ? 8 : 0;
#pragma unroll
    for (int r = 0; r < 8; ++r) {
      const int kk = ti * 16 + r + mofs;
      const int nn = tj * 16 + n;
      outc[kk * DHEAD + nn] = su[r];
    }
  }
}

// ---------------------------------------------------------------------------
// Kernel 5b: exclusive prefix over chunks per (b,h): S_exc[c] = sum_{c'<c} KV[c'].
// Emitted in bf16, TRANSPOSED ([n][k]) so pass 5c can load it directly as a
// WMMA B-operand from global memory.
// ---------------------------------------------------------------------------
__global__ __launch_bounds__(256) void scan_kv(const float* __restrict__ KVc,
                                               bf16_t* __restrict__ Sbtg) {
  const int bh = blockIdx.x;                // 32 blocks
  const int tid = threadIdx.x;
  float run[16];
#pragma unroll
  for (int t = 0; t < 16; ++t) run[t] = 0.f;

  for (int c = 0; c < NCHUNK; ++c) {
    const size_t cb = ((size_t)bh * NCHUNK + c) * DHEAD * DHEAD;
#pragma unroll
    for (int t = 0; t < 16; ++t) {
      const int i = tid + t * 256;          // (k,n) flat, k-major
      const int k = i >> 6, n = i & 63;
      Sbtg[cb + n * DHEAD + k] = (bf16_t)run[t];   // exclusive, transposed
      run[t] += KVc[cb + i];
    }
  }
}

// ---------------------------------------------------------------------------
// Kernel 5c: per-chunk output (fully parallel over B*H*NCHUNK blocks).
//   P = tril(Qc @ Kc^T);  Y = Qc @ S_exc + P @ Vc
// Q chunk staged row-major into LDS via GLOBAL_LOAD_ASYNC_TO_LDS_B128
// (reused by both phases); S_exc B-operand streamed straight from global
// (already transposed bf16).
// ---------------------------------------------------------------------------
__global__ __launch_bounds__(256) void attn_out(const bf16_t* __restrict__ Qg,
                                                const bf16_t* __restrict__ Kg,
                                                const bf16_t* __restrict__ Vg,
                                                const bf16_t* __restrict__ Sbtg,
                                                bf16_t* __restrict__ Yg) {
  __shared__ bf16_t Qs[DHEAD * DHEAD];      // Q chunk, row-major [i][kd]
  __shared__ bf16_t Pb[DHEAD * DHEAD];      // masked P, row-major [i][j]
  __shared__ bf16_t Vt[DHEAD * DHEAD];      // Vt[n][j] = V[l0+j][n]

  const int bhc = blockIdx.x;
  const int bh = bhc / NCHUNK;
  const int c = bhc % NCHUNK;
  const int bb = bh / NHEAD, h = bh % NHEAD;
  const int l0 = c * 64;
  const size_t base = (size_t)bh * LSEQ * DHEAD;
  const bf16_t* Q = Qg + base;
  const bf16_t* Km = Kg + base;
  const bf16_t* V = Vg + base;
  const bf16_t* Sc = Sbtg + (size_t)bhc * DHEAD * DHEAD;   // [n][k] bf16

  const int tid = threadIdx.x;
  const int lane = tid & 31;
  const int wave = tid >> 5;

  // Stage Q chunk row-major via async DMA: 8 KB = 512 x 16B, 2 per thread.
  {
    const bf16_t* qsrc = Q + (size_t)l0 * DHEAD;
#pragma unroll
    for (int s = 0; s < 2; ++s) {
      const int seg = tid + s * 256;        // 0..511
      async_copy_b128(qsrc + seg * 8, &Qs[seg * 8]);
    }
  }
  // Transposed V staging (scalar; transpose not expressible as straight copy).
  for (int i = tid; i < DHEAD * DHEAD; i += 256) {
    const int row = i >> 6, col = i & 63;
    Vt[i] = V[(size_t)(l0 + col) * DHEAD + row];
  }
  wait_async0();
  __syncthreads();

  // Phase 1: P = tril(Qc @ Kc^T)
  for (int t = wave; t < 16; t += 8) {
    const int ti = t >> 2, tj = t & 3;
    v8f acc = {};
    for (int k = 0; k < DHEAD; k += 32) {
      v16bf a = load_A_rm(Qs + (size_t)(ti * 16) * DHEAD + k, DHEAD, lane);
      v16bf b = load_B_t(Km + (size_t)(l0 + tj * 16) * DHEAD + k, DHEAD, lane);
      acc = wmma_bf16(a, b, acc);
    }
    const int n = lane & 15;
    const int mofs = (lane >> 4) ? 8 : 0;
#pragma unroll
    for (int r = 0; r < 8; ++r) {
      const int i = ti * 16 + r + mofs;
      const int j = tj * 16 + n;
      Pb[i * DHEAD + j] = (bf16_t)(j <= i ? acc[r] : 0.f);
    }
  }
  __syncthreads();

  // Phase 2: Y = Qc @ S_exc + P @ Vc
  for (int t = wave; t < 16; t += 8) {
    const int ti = t >> 2, tj = t & 3;
    v8f acc = {};
    for (int k = 0; k < DHEAD; k += 32) {
      v16bf a = load_A_rm(Qs + (size_t)(ti * 16) * DHEAD + k, DHEAD, lane);
      v16bf b = load_B_t(Sc + (size_t)(tj * 16) * DHEAD + k, DHEAD, lane);
      acc = wmma_bf16(a, b, acc);
      v16bf a2 = load_A_rm(Pb + (size_t)(ti * 16) * DHEAD + k, DHEAD, lane);
      v16bf b2 = load_B_t(Vt + (size_t)(tj * 16) * DHEAD + k, DHEAD, lane);
      acc = wmma_bf16(a2, b2, acc);
    }
    const int n = lane & 15;
    const int mofs = (lane >> 4) ? 8 : 0;
#pragma unroll
    for (int r = 0; r < 8; ++r) {
      const int i = l0 + ti * 16 + r + mofs;            // sequence position
      const int j = tj * 16 + n;                        // head dim
      Yg[((size_t)(bb * LSEQ + i)) * DMODEL + h * DHEAD + j] = (bf16_t)acc[r];
    }
  }
}

// ---------------------------------------------------------------------------
// Host-side launch
// ---------------------------------------------------------------------------
extern "C" void kernel_launch(void* const* d_in, const int* in_sizes, int n_in,
                              void* d_out, int out_size, void* d_ws, size_t ws_size,
                              hipStream_t stream) {
  (void)in_sizes; (void)n_in; (void)out_size; (void)ws_size;
  const float* X    = (const float*)d_in[0];
  const float* Wq   = (const float*)d_in[1];
  const float* Wk   = (const float*)d_in[2];
  const float* Wv   = (const float*)d_in[3];
  const float* Wc   = (const float*)d_in[4];
  const float* Ltri = (const float*)d_in[5];
  float* out = (float*)d_out;

  // Workspace carve-up (bytes)
  char* ws = (char*)d_ws;
  size_t off = 0;
  auto carve = [&](size_t bytes) { char* p = ws + off; off += (bytes + 255) & ~(size_t)255; return p; };
  bf16_t* Xb   = (bf16_t*)carve((size_t)MTOT * DMODEL * 2);      // 8 MB
  bf16_t* Wqb  = (bf16_t*)carve((size_t)DMODEL * DMODEL * 2);    // 2 MB
  bf16_t* Wkb  = (bf16_t*)carve((size_t)DMODEL * DMODEL * 2);
  bf16_t* Wvb  = (bf16_t*)carve((size_t)DMODEL * DMODEL * 2);
  bf16_t* Wc2b = (bf16_t*)carve((size_t)DMODEL * DMODEL * 2);
  bf16_t* Qb   = (bf16_t*)carve((size_t)MTOT * DMODEL * 2);      // [B,H,L,d]
  bf16_t* Kb   = (bf16_t*)carve((size_t)MTOT * DMODEL * 2);
  bf16_t* Vb   = (bf16_t*)carve((size_t)MTOT * DMODEL * 2);
  bf16_t* Yb   = (bf16_t*)carve((size_t)MTOT * DMODEL * 2);      // [m][D]
  float*  Bm   = (float*)carve((size_t)NHEAD * DHEAD * DHEAD * 4);
  float*  KVc  = (float*)carve((size_t)NBH * NCHUNK * DHEAD * DHEAD * 4);   // 16 MB
  bf16_t* Sbtg = (bf16_t*)carve((size_t)NBH * NCHUNK * DHEAD * DHEAD * 2);  //  8 MB

  // 1) casts
  {
    const int nX = MTOT * DMODEL;
    cast_f32_bf16<<<(nX + 255) / 256, 256, 0, stream>>>(X, Xb, nX);
    const int nW = DMODEL * DMODEL;
    cast_f32_bf16<<<(nW + 255) / 256, 256, 0, stream>>>(Wq, Wqb, nW);
    cast_f32_bf16<<<(nW + 255) / 256, 256, 0, stream>>>(Wk, Wkb, nW);
    cast_f32_bf16<<<(nW + 255) / 256, 256, 0, stream>>>(Wv, Wvb, nW);
  }

  // 2) head-mix matrix + fold into Wc
  build_bm<<<NHEAD, 256, 0, stream>>>(Ltri, Bm);
  fold_wc<<<(DMODEL * DMODEL + 255) / 256, 256, 0, stream>>>(Wc, Bm, Wc2b);

  // 3) Q/K/V projections (pipelined WMMA GEMM, TDM LDS staging)
  const int gemm_blocks = (MTOT / 128) * (DMODEL / 64);          // 512
  gemm_wmma<true><<<gemm_blocks, 256, 0, stream>>>(Xb, Wqb, (void*)Qb);
  gemm_wmma<true><<<gemm_blocks, 256, 0, stream>>>(Xb, Wkb, (void*)Kb);
  gemm_wmma<true><<<gemm_blocks, 256, 0, stream>>>(Xb, Wvb, (void*)Vb);

  // 4) causal linear attention: parallel chunk KV -> scan -> parallel output
  chunk_kv<<<NBH * NCHUNK, 256, 0, stream>>>(Kb, Vb, KVc);
  scan_kv<<<NBH, 256, 0, stream>>>(KVc, Sbtg);
  attn_out<<<NBH * NCHUNK, 256, 0, stream>>>(Qb, Kb, Vb, Sbtg, Yb);

  // 5) output projection with folded head-mix (f32 out)
  gemm_wmma<false><<<gemm_blocks, 256, 0, stream>>>(Yb, Wc2b, (void*)out);
}
